// PC_MO_LSTM_NOC_5454608466687
// MI455X (gfx1250) — compile-verified, matrix operand order checked by reading, not run on hardware
//
#include <hip/hip_runtime.h>
#include <hip/hip_fp16.h>

// ---------------------------------------------------------------------------
// CDNA5 (gfx1250) point-cloud LSTM primitives.
// GEMMs run through v_wmma_f32_16x16x32_f16 (f16 in, f32 acc), 64x64 block
// tile, 8 waves (4Mx2N), 2 accumulators/wave, double-buffered LDS with
// async global->LDS DMA (global_load_async_to_lds_b128 + s_wait_asynccnt).
// B is stored transposed in LDS so both fragments load as 2x ds_load_b128.
// ---------------------------------------------------------------------------

typedef __attribute__((ext_vector_type(16))) _Float16 v16h;
typedef __attribute__((ext_vector_type(8)))  _Float16 v8h;
typedef __attribute__((ext_vector_type(8)))  float    v8f;
typedef __attribute__((ext_vector_type(4)))  int      v4i;

#ifndef __has_builtin
#define __has_builtin(x) 0
#endif

#if __has_builtin(__builtin_amdgcn_global_load_async_to_lds_b128) && \
    __has_builtin(__builtin_amdgcn_s_wait_asynccnt)
#define USE_ASYNC_LDS 1
#else
#define USE_ASYNC_LDS 0
#endif

static __device__ __forceinline__ int tid1d() {
  return blockIdx.x * blockDim.x + threadIdx.x;
}

// ---------------------------------------------------------------------------
// [B,3,N] frame -> [B,N,3]
__global__ void k_transpose_xyz(const float* __restrict__ src,
                                float* __restrict__ dst, int B, int N) {
  int i = tid1d();
  if (i >= B * N) return;
  int b = i / N, n = i % N;
  dst[(size_t)i * 3 + 0] = src[((size_t)b * 3 + 0) * N + n];
  dst[(size_t)i * 3 + 1] = src[((size_t)b * 3 + 1) * N + n];
  dst[(size_t)i * 3 + 2] = src[((size_t)b * 3 + 2) * N + n];
}

// ---------------------------------------------------------------------------
// Farthest point sampling. One 256-thread block per batch element.
__global__ __launch_bounds__(256)
void k_fps(const float* __restrict__ pos, int* __restrict__ out_idx,
           float* __restrict__ dscratch, int N, int n) {
  int b = blockIdx.x;
  const float* P = pos + (size_t)b * N * 3;
  float* dmin = dscratch + (size_t)b * N;
  __shared__ float s_best[256];
  __shared__ int   s_besti[256];
  __shared__ int   s_last;
  int t = threadIdx.x;
  for (int i = t; i < N; i += 256) dmin[i] = 1e10f;
  if (t == 0) s_last = 0;
  __syncthreads();
  for (int it = 0; it < n; ++it) {
    int last = s_last;
    if (t == 0) out_idx[(size_t)b * n + it] = last;
    float lx = P[(size_t)last * 3 + 0];
    float ly = P[(size_t)last * 3 + 1];
    float lz = P[(size_t)last * 3 + 2];
    float best = -1.0f; int besti = 0;
    for (int i = t; i < N; i += 256) {
      float dx = P[(size_t)i * 3 + 0] - lx;
      float dy = P[(size_t)i * 3 + 1] - ly;
      float dz = P[(size_t)i * 3 + 2] - lz;
      float d  = dx * dx + dy * dy + dz * dz;
      float dm = fminf(dmin[i], d);
      dmin[i] = dm;
      if (dm > best) { best = dm; besti = i; }
    }
    s_best[t] = best; s_besti[t] = besti;
    __syncthreads();
    for (int off = 128; off > 0; off >>= 1) {
      if (t < off && s_best[t + off] > s_best[t]) {
        s_best[t] = s_best[t + off]; s_besti[t] = s_besti[t + off];
      }
      __syncthreads();
    }
    if (t == 0) s_last = s_besti[0];
    __syncthreads();
  }
}

// ---------------------------------------------------------------------------
__global__ void k_gather_pos(const float* __restrict__ pos,
                             const int* __restrict__ idx,
                             float* __restrict__ out, int B, int N, int n) {
  int i = tid1d();
  if (i >= B * n) return;
  int b = i / n;
  int s = idx[i];
  const float* p = pos + ((size_t)b * N + s) * 3;
  out[(size_t)i * 3 + 0] = p[0];
  out[(size_t)i * 3 + 1] = p[1];
  out[(size_t)i * 3 + 2] = p[2];
}

// ---------------------------------------------------------------------------
// Register-resident kNN: unrolled bubble insert, no scratch, one thread/query.
template <int K>
__global__ void k_knn(const float* __restrict__ q, const float* __restrict__ ref,
                      int* __restrict__ nidx, float* __restrict__ ndist,
                      int B, int Q, int R) {
  int i = tid1d();
  if (i >= B * Q) return;
  int b = i / Q;
  const float* Rp = ref + (size_t)b * R * 3;
  float qx = q[(size_t)i * 3 + 0];
  float qy = q[(size_t)i * 3 + 1];
  float qz = q[(size_t)i * 3 + 2];
  float bd[K]; int bi[K];
#pragma unroll
  for (int j = 0; j < K; ++j) { bd[j] = 1e30f; bi[j] = 0; }
  for (int r = 0; r < R; ++r) {
    float dx = Rp[(size_t)r * 3 + 0] - qx;
    float dy = Rp[(size_t)r * 3 + 1] - qy;
    float dz = Rp[(size_t)r * 3 + 2] - qz;
    float cd = dx * dx + dy * dy + dz * dz;
    int ci = r;
#pragma unroll
    for (int j = 0; j < K; ++j) {
      bool c = cd < bd[j];
      float td = bd[j]; int ti = bi[j];
      bd[j] = c ? cd : td; bi[j] = c ? ci : ti;
      cd = c ? td : cd;    ci = c ? ti : ci;
    }
  }
#pragma unroll
  for (int j = 0; j < K; ++j) {
    nidx[(size_t)i * K + j] = bi[j];
    if (ndist) ndist[(size_t)i * K + j] = bd[j];
  }
}

// ---------------------------------------------------------------------------
// sa1 grouping: [abs_xyz, rel_xyz] -> f16 row, zero-padded to Cpad.
__global__ void k_group_sa1(const float* __restrict__ pos,
                            const float* __restrict__ newpos,
                            const int* __restrict__ nidx,
                            _Float16* __restrict__ out,
                            int B, int N, int n, int k, int Cpad) {
  int i = tid1d();
  if (i >= B * n * k) return;
  int row = i / k;           // b*n + q
  int b = row / n;
  int s = nidx[i];
  const float* p = pos + ((size_t)b * N + s) * 3;
  const float* c = newpos + (size_t)row * 3;
  _Float16* o = out + (size_t)i * Cpad;
  o[0] = (_Float16)p[0]; o[1] = (_Float16)p[1]; o[2] = (_Float16)p[2];
  o[3] = (_Float16)(p[0] - c[0]);
  o[4] = (_Float16)(p[1] - c[1]);
  o[5] = (_Float16)(p[2] - c[2]);
  for (int j = 6; j < Cpad; ++j) o[j] = (_Float16)0.0f;
}

// ---------------------------------------------------------------------------
// Pack fp32 weight [Kin,Cout] into zero-padded f16 [Kpad,Npad].
__global__ void k_pack_w(const float* __restrict__ W, _Float16* __restrict__ out,
                         int Kin, int Cout, int Kpad, int Npad) {
  int i = tid1d();
  if (i >= Kpad * Npad) return;
  int r = i / Npad, c = i % Npad;
  out[i] = (r < Kin && c < Cout) ? (_Float16)W[(size_t)r * Cout + c]
                                 : (_Float16)0.0f;
}

// ---------------------------------------------------------------------------
// f32 -> f16 with column zero-pad (cls head inter-layer repack).
__global__ void k_cvt_pad(const float* __restrict__ X, _Float16* __restrict__ Y,
                          int rows, int cols, int Cpad) {
  int i = tid1d();
  if (i >= rows * Cpad) return;
  int r = i / Cpad, c = i % Cpad;
  Y[i] = (c < cols) ? (_Float16)X[(size_t)r * cols + c] : (_Float16)0.0f;
}

// ---------------------------------------------------------------------------
// WMMA GEMM: C[M,N] = A[M,K] * B[K,N]; A,B f16 row-major (K%32==0, N%16==0),
// C f32. 256 threads = 8 waves tiled 4(M) x 2(N) over a 64x64 block tile;
// each wave owns two 16x16 f32 accumulators (A fragment reused).
// Double-buffered LDS; A staged via async global->LDS b128 DMA; B staged
// transposed (sBt[col][k]) so fragments load as contiguous ds_load_b128.
__global__ __launch_bounds__(256)
void k_gemm_wmma(const _Float16* __restrict__ A, const _Float16* __restrict__ Bm,
                 float* __restrict__ C, int M, int N, int K) {
  __shared__ _Float16 sA [2][64][48];  // [buf][row m][k], 96B rows (16B aligned)
  __shared__ _Float16 sBt[2][64][48];  // [buf][col n][k], transposed B

  const int t     = threadIdx.x;
  const int lane  = t & 31;
  const int wv    = t >> 5;
  const int wm    = (wv & 3) * 16;   // wave M offset in block tile
  const int wn    = (wv >> 2) * 32;  // wave N offset (covers 2x16 cols)
  const int bM    = blockIdx.x * 64;
  const int bN    = blockIdx.y * 64;
  const int lrow  = lane & 15;
  const int lhalf = lane >> 4;       // which K-half this lane holds

  // cooperative-load coordinates
  const int ar = t >> 2;             // A: row 0..63
  const int ac = (t & 3) * 8;        // A: k group {0,8,16,24}
  const int brr = t >> 3;            // B: k row 0..31
  const int bcc = (t & 7) * 8;       // B: col group {0..56}

  auto load_tile = [&](int buf, int k0) {
    // ---- A tile: 64x32, one b128 per thread, async DMA into LDS ----
    int gm = bM + ar;
    const _Float16* srcA = A + (size_t)gm * K + (k0 + ac);
    if (gm < M) {
#if USE_ASYNC_LDS
      __builtin_amdgcn_global_load_async_to_lds_b128(
          (v4i*)srcA,
          (__attribute__((address_space(3))) v4i*)&sA[buf][ar][ac], 0, 0);
#else
      *(v8h*)&sA[buf][ar][ac] = *(const v8h*)srcA;
#endif
      if (k0 + 32 < K)
        __builtin_prefetch(A + (size_t)gm * K + (k0 + 32 + ac), 0, 1);
    } else {
      v8h z = {};
      *(v8h*)&sA[buf][ar][ac] = z;
    }
    // ---- B tile: 32x64 row-major in memory -> transposed scatter ----
    {
      int gn = bN + bcc;
      const _Float16* srcB = Bm + (size_t)(k0 + brr) * N + gn;
      if (gn + 7 < N) {
        v8h v = *(const v8h*)srcB;
#pragma unroll
        for (int j = 0; j < 8; ++j) sBt[buf][bcc + j][brr] = v[j];
      } else {
#pragma unroll
        for (int j = 0; j < 8; ++j)
          sBt[buf][bcc + j][brr] = (gn + j < N) ? srcB[j] : (_Float16)0.0f;
      }
    }
  };

  v8f acc0 = {}, acc1 = {};
  const int nk = K >> 5;

  load_tile(0, 0);
  for (int kt = 0; kt < nk; ++kt) {
#if USE_ASYNC_LDS
    __builtin_amdgcn_s_wait_asynccnt(0);
#endif
    __syncthreads();
    const int buf = kt & 1;
    if (kt + 1 < nk) load_tile(buf ^ 1, (kt + 1) << 5);

    // CDNA5 16-bit fragment layout: lanes 0-15 hold K {0..7,16..23},
    // lanes 16-31 hold K {8..15,24..31}; both segments contiguous in LDS.
    v16h af, bf0, bf1;
#pragma unroll
    for (int e = 0; e < 16; ++e) {
      const int kk = e + lhalf * 8 + (e & 8);
      af[e]  = sA [buf][wm + lrow][kk];
      bf0[e] = sBt[buf][wn + lrow][kk];
      bf1[e] = sBt[buf][wn + 16 + lrow][kk];
    }
    acc0 = __builtin_amdgcn_wmma_f32_16x16x32_f16(
        false, af, false, bf0, (short)0, acc0, false, false);
    acc1 = __builtin_amdgcn_wmma_f32_16x16x32_f16(
        false, af, false, bf1, (short)0, acc1, false, false);
  }

  // C layout: VGPR r, lanes 0-15 -> M=r, lanes 16-31 -> M=r+8; N=lane&15.
#pragma unroll
  for (int r = 0; r < 8; ++r) {
    int gm = bM + wm + lhalf * 8 + r;
    int gn = bN + wn + lrow;
    if (gm < M) {
      if (gn < N)      C[(size_t)gm * N + gn]      = acc0[r];
      if (gn + 16 < N) C[(size_t)gm * N + gn + 16] = acc1[r];
    }
  }
}

// ---------------------------------------------------------------------------
// Per-column batch statistics (mean, biased var). One block per column.
__global__ __launch_bounds__(256)
void k_col_stats(const float* __restrict__ X, int rows, int cols,
                 float* __restrict__ mean, float* __restrict__ var) {
  int c = blockIdx.x;
  float s = 0.0f, s2 = 0.0f;
  for (int r = threadIdx.x; r < rows; r += 256) {
    float v = X[(size_t)r * cols + c];
    s += v; s2 += v * v;
  }
  __shared__ float sh[256], sh2[256];
  sh[threadIdx.x] = s; sh2[threadIdx.x] = s2;
  __syncthreads();
  for (int o = 128; o > 0; o >>= 1) {
    if (threadIdx.x < o) {
      sh[threadIdx.x]  += sh[threadIdx.x + o];
      sh2[threadIdx.x] += sh2[threadIdx.x + o];
    }
    __syncthreads();
  }
  if (threadIdx.x == 0) {
    float mu = sh[0] / (float)rows;
    mean[c] = mu;
    var[c] = sh2[0] / (float)rows - mu * mu;
  }
}

// ---------------------------------------------------------------------------
// Fused BN + ReLU; writes f16 (padded) for the next GEMM, optional f32 copy.
__global__ void k_bn_relu(const float* __restrict__ X,
                          const float* __restrict__ mean,
                          const float* __restrict__ var,
                          const float* __restrict__ g,
                          const float* __restrict__ b,
                          _Float16* __restrict__ Y, float* __restrict__ Yf,
                          int rows, int cols, int Cpad) {
  int i = tid1d();
  if (i >= rows * Cpad) return;
  int r = i / Cpad, c = i % Cpad;
  float y = 0.0f;
  if (c < cols) {
    float x = X[(size_t)r * cols + c];
    y = (x - mean[c]) * rsqrtf(var[c] + 1e-5f) * g[c] + b[c];
    y = fmaxf(y, 0.0f);
    if (Yf) Yf[(size_t)r * cols + c] = y;
  }
  Y[i] = (_Float16)y;
}

// ---------------------------------------------------------------------------
// Max-pool over the neighbor axis: [groups,k,C] -> [groups,C].
__global__ void k_maxpool(const float* __restrict__ X, float* __restrict__ Y,
                          int groups, int k, int C) {
  int i = tid1d();
  if (i >= groups * C) return;
  int g = i / C, c = i % C;
  const float* p = X + (size_t)g * k * C + c;
  float m = -1e30f;
  for (int j = 0; j < k; ++j) m = fmaxf(m, p[(size_t)j * C]);
  Y[i] = m;
}

// ---------------------------------------------------------------------------
// Inverse-distance-weighted kNN interpolation -> f16 (padded).
__global__ void k_interp(const float* __restrict__ F, const int* __restrict__ nidx,
                         const float* __restrict__ nd, _Float16* __restrict__ Y,
                         int B, int n, int Nf, int k, int C, int Cpad) {
  int i = tid1d();
  if (i >= B * Nf * Cpad) return;
  int c = i % Cpad;
  int row = i / Cpad;
  int b = row / Nf;
  float y = 0.0f;
  if (c < C) {
    const int* id = nidx + (size_t)row * k;
    const float* dd = nd + (size_t)row * k;
    float ws = 0.0f;
    for (int j = 0; j < k; ++j) {
      float w = 1.0f / (dd[j] + 1e-8f);
      ws += w;
      y += w * F[((size_t)b * n + id[j]) * C + c];
    }
    y /= ws;
  }
  Y[i] = (_Float16)y;
}

// ---------------------------------------------------------------------------
// out[p,b,n,:] = pc + offsets[:, 0:3] for each of num_pred predictions.
__global__ void k_residual_out(const float* __restrict__ pc,
                               const float* __restrict__ off,
                               float* __restrict__ out, int total3, int Np,
                               int num_pred) {
  int i = tid1d();
  if (i >= total3) return;
  int row = i / 3, c = i % 3;
  float v = pc[i] + off[(size_t)row * Np + c];
  for (int p = 0; p < num_pred; ++p) out[(size_t)p * total3 + i] = v;
}

// ---------------------------------------------------------------------------
extern "C" void kernel_launch(void* const* d_in, const int* in_sizes, int n_in,
                              void* d_out, int out_size, void* d_ws,
                              size_t ws_size, hipStream_t stream) {
  (void)in_sizes; (void)out_size; (void)ws_size;
  const int T = 4, B = 2, N = 8192;
  const int n1 = N / 16;            // 512
  const int K1 = 32;                // sa1 neighborhood
  const int KI = 16;                // interpolation kNN
  const int rows1 = B * n1 * K1;    // 32768
  const int rowsN = B * N;          // 16384
  const int num_pred = 2;

  const float* xyz = (const float*)d_in[0];
  const float* frame_last = xyz + (size_t)(T - 1) * B * 3 * N;

  // param layout per setup_inputs() flattening order
  const float* saW[3] = {(const float*)d_in[1], (const float*)d_in[4], (const float*)d_in[7]};
  const float* saG[3] = {(const float*)d_in[2], (const float*)d_in[5], (const float*)d_in[8]};
  const float* saBb[3] = {(const float*)d_in[3], (const float*)d_in[6], (const float*)d_in[9]};
  int c0 = n_in - 5; if (c0 < 1) c0 = 1;
  const float* clsW[4] = {(const float*)d_in[c0], (const float*)d_in[c0 + 1],
                          (const float*)d_in[c0 + 2], (const float*)d_in[c0 + 3]};

  // workspace carve (bump allocator, 256B aligned)
  char* w = (char*)d_ws;
  auto alloc = [&](size_t bytes) -> char* {
    char* p = w;
    w += (bytes + 255) & ~(size_t)255;
    return p;
  };
  float*    pc     = (float*)alloc((size_t)rowsN * 3 * 4);
  float*    dsc    = (float*)alloc((size_t)B * N * 4);
  int*      fpsi   = (int*)alloc((size_t)B * n1 * 4);
  float*    npos   = (float*)alloc((size_t)B * n1 * 3 * 4);
  int*      nidx1  = (int*)alloc((size_t)rows1 * 4);
  _Float16* gin    = (_Float16*)alloc((size_t)rowsN * 128 * 2);
  float*    act    = (float*)alloc((size_t)rowsN * 128 * 4);
  _Float16* wp     = (_Float16*)alloc((size_t)128 * 128 * 2);
  float*    mean   = (float*)alloc(512 * 4);
  float*    var    = (float*)alloc(512 * 4);
  float*    f1     = (float*)alloc((size_t)B * n1 * 64 * 4);
  int*      nidxN  = (int*)alloc((size_t)rowsN * KI * 4);
  float*    ndistN = (float*)alloc((size_t)rowsN * KI * 4);

  dim3 blk(256);
  auto cdiv = [](int a, int b) { return (a + b - 1) / b; };

  // ---- encode (last frame): FPS -> kNN -> group -> MLP(WMMA)+BN+ReLU ----
  k_transpose_xyz<<<cdiv(rowsN, 256), blk, 0, stream>>>(frame_last, pc, B, N);
  k_fps<<<B, blk, 0, stream>>>(pc, fpsi, dsc, N, n1);
  k_gather_pos<<<cdiv(B * n1, 256), blk, 0, stream>>>(pc, fpsi, npos, B, N, n1);
  k_knn<32><<<cdiv(B * n1, 256), blk, 0, stream>>>(npos, pc, nidx1,
                                                   (float*)nullptr, B, n1, N);
  k_group_sa1<<<cdiv(rows1, 256), blk, 0, stream>>>(pc, npos, nidx1, gin,
                                                    B, N, n1, K1, 32);

  const int din[3]  = {6, 32, 32};
  const int dout[3] = {32, 32, 64};
  for (int l = 0; l < 3; ++l) {
    const int Kp = 32, Nn = dout[l], Cp = Nn;  // Nn in {32,64}: already mult 32
    k_pack_w<<<cdiv(Kp * Nn, 256), blk, 0, stream>>>(saW[l], wp, din[l], Nn, Kp, Nn);
    dim3 g(cdiv(rows1, 64), cdiv(Nn, 64));
    k_gemm_wmma<<<g, blk, 0, stream>>>(gin, wp, act, rows1, Nn, Kp);
    k_col_stats<<<Nn, blk, 0, stream>>>(act, rows1, Nn, mean, var);
    k_bn_relu<<<cdiv(rows1 * Cp, 256), blk, 0, stream>>>(
        act, mean, var, saG[l], saBb[l], gin, (l == 2) ? act : (float*)nullptr,
        rows1, Nn, Cp);
  }
  k_maxpool<<<cdiv(B * n1 * 64, 256), blk, 0, stream>>>(act, f1, B * n1, K1, 64);

  // ---- decode: IDW interpolation up to full resolution, cls head ----
  k_knn<16><<<cdiv(rowsN, 256), blk, 0, stream>>>(pc, npos, nidxN, ndistN,
                                                  B, N, n1);
  k_interp<<<cdiv(rowsN * 128, 256), blk, 0, stream>>>(f1, nidxN, ndistN, gin,
                                                       B, n1, N, KI, 64, 128);

  const int ck[4]  = {128, 128, 64, 16};  // true input dims
  const int cn[4]  = {128, 64, 16, 3};    // true output dims
  const int ckp[4] = {128, 128, 64, 32};  // K padded to 32
  const int cnp[4] = {128, 64, 16, 16};   // N padded to 16
  for (int l = 0; l < 4; ++l) {
    k_pack_w<<<cdiv(ckp[l] * cnp[l], 256), blk, 0, stream>>>(
        clsW[l], wp, ck[l], cn[l], ckp[l], cnp[l]);
    dim3 g(cdiv(rowsN, 64), cdiv(cnp[l], 64));
    k_gemm_wmma<<<g, blk, 0, stream>>>(gin, wp, act, rowsN, cnp[l], ckp[l]);
    if (l < 3)
      k_cvt_pad<<<cdiv(rowsN * ckp[l + 1], 256), blk, 0, stream>>>(
          act, gin, rowsN, cn[l], ckp[l + 1]);
  }

  k_residual_out<<<cdiv(rowsN * 3, 256), blk, 0, stream>>>(
      pc, act, (float*)d_out, rowsN * 3, cnp[3], num_pred);
}
